// CharLSTMTagger_10187662426745
// MI455X (gfx1250) — matrix operand order repaired and checked by reading, not compiled, not run on hardware
//
#include <hip/hip_runtime.h>
#include <hip/hip_bf16.h>
#include <stdint.h>

// ---------------------------------------------------------------------------
// CharLSTM tagger for MI455X (gfx1250).
//  - Char LSTM: 16 dependent bf16 WMMA GEMMs (M=2048,N=1024,K=512) + fused cell.
//  - Word LSTM input projection: one bf16 WMMA GEMM (M=2048,N=4096,K=768).
//  - Word LSTM recurrence (batch=1): persistent 8-block kernel, bf16 weights
//    (L2-resident), device-wide barrier per step.
//  - GEMM stages A/B tiles to LDS with gfx1250 async load-to-LDS (ASYNCcnt),
//    double-buffered; falls back to sync staging if builtins are absent.
// ---------------------------------------------------------------------------

typedef __bf16 bf16_t;
typedef __attribute__((ext_vector_type(8)))  bf16_t bf16x8;
typedef __attribute__((ext_vector_type(16))) bf16_t v16bf;
typedef __attribute__((ext_vector_type(8)))  float  v8f;
typedef __attribute__((ext_vector_type(4)))  int    v4i;

#if __has_builtin(__builtin_amdgcn_global_load_async_to_lds_b128)
#define USE_ASYNC_LDS 1
#endif

static constexpr int S      = 2048;
static constexpr int LMAX   = 16;
static constexpr int CE     = 256;
static constexpr int WE     = 512;
static constexpr int HDIM   = 1024;
static constexpr int KC     = 2 * CE;        // 512  char GEMM K
static constexpr int NC     = 4 * CE;        // 1024 char gate count
static constexpr int KW     = WE + CE;       // 768  word input K
static constexpr int NW     = 4 * HDIM;      // 4096 word gate count

// ---- workspace layout (all offsets 256B aligned) --------------------------
static constexpr size_t OFF_HBF   = 0;                               // bf16 [S][CE] char h
static constexpr size_t SZ_HBF    = (size_t)S * CE * 2;
static constexpr size_t OFF_CC    = OFF_HBF + SZ_HBF;                // f32  [S][CE] char c
static constexpr size_t SZ_CC     = (size_t)S * CE * 4;
static constexpr size_t OFF_HW    = OFF_CC + SZ_CC;                  // bf16 [HDIM] word h
static constexpr size_t SZ_HW     = (size_t)HDIM * 2;
static constexpr size_t OFF_CTR   = OFF_HW + ((SZ_HW + 255) & ~(size_t)255);
static constexpr size_t SZ_CTR    = 256;                             // barrier counters
static constexpr size_t ZERO_BYTES= OFF_CTR + SZ_CTR;                // zeroed every launch

static constexpr size_t OFF_E     = ZERO_BYTES;                      // bf16 [LMAX][S][CE]
static constexpr size_t SZ_E      = (size_t)LMAX * S * CE * 2;
static constexpr size_t OFF_WC    = OFF_E + SZ_E;                    // bf16 [NC][KC]
static constexpr size_t SZ_WC     = (size_t)NC * KC * 2;
static constexpr size_t OFF_BC    = OFF_WC + SZ_WC;                  // f32 [NC]
static constexpr size_t SZ_BC     = (size_t)NC * 4;
static constexpr size_t OFF_LAST  = OFF_BC + SZ_BC;                  // f32 [S][CE]
static constexpr size_t SZ_LAST   = (size_t)S * CE * 4;
static constexpr size_t OFF_GATES = OFF_LAST + SZ_LAST;              // f32 [S][NC]
static constexpr size_t SZ_GATES  = (size_t)S * NC * 4;
static constexpr size_t OFF_XW    = OFF_GATES + SZ_GATES;            // bf16 [S][KW]
static constexpr size_t SZ_XW     = (size_t)S * KW * 2;
static constexpr size_t OFF_WIHW  = OFF_XW + SZ_XW;                  // bf16 [NW][KW]
static constexpr size_t SZ_WIHW   = (size_t)NW * KW * 2;
static constexpr size_t OFF_WHHW  = OFF_WIHW + SZ_WIHW;              // bf16 [NW][HDIM]
static constexpr size_t SZ_WHHW   = (size_t)NW * HDIM * 2;
static constexpr size_t OFF_BW    = OFF_WHHW + SZ_WHHW;              // f32 [NW]
static constexpr size_t SZ_BW     = (size_t)NW * 4;
static constexpr size_t OFF_GIN   = OFF_BW + SZ_BW;                  // f32 [S][NW]

__device__ __forceinline__ float sigm(float x) { return 1.0f / (1.0f + __expf(-x)); }

__device__ __forceinline__ v16bf comb16(bf16x8 lo, bf16x8 hi) {
    v16bf r;
#pragma unroll
    for (int i = 0; i < 8; ++i) { r[i] = lo[i]; r[8 + i] = hi[i]; }
    return r;
}

// 16B global -> LDS copy; async (ASYNCcnt-tracked) when toolchain exposes it.
__device__ __forceinline__ void stage16(const bf16_t* g, bf16_t* l) {
#ifdef USE_ASYNC_LDS
    __builtin_amdgcn_global_load_async_to_lds_b128(
        (__attribute__((address_space(1))) v4i*)g,
        (__attribute__((address_space(3))) v4i*)l, 0, 0);
#else
    *(bf16x8*)l = *(const bf16x8*)g;
#endif
}

__device__ __forceinline__ void wait_stage() {
#if __has_builtin(__builtin_amdgcn_s_wait_asynccnt)
    __builtin_amdgcn_s_wait_asynccnt(0);
#elif defined(USE_ASYNC_LDS)
    asm volatile("s_wait_asynccnt 0x0" ::: "memory");
#endif
    __syncthreads();
}

// ---------------------------------------------------------------------------
// prep kernels
// ---------------------------------------------------------------------------
__global__ void zero_ws_kernel(uint4* p, size_t n16) {
    for (size_t i = (size_t)blockIdx.x * blockDim.x + threadIdx.x; i < n16;
         i += (size_t)gridDim.x * blockDim.x)
        p[i] = make_uint4(0u, 0u, 0u, 0u);
}

__global__ void f32_to_bf16_kernel(const float* __restrict__ src, bf16_t* __restrict__ dst,
                                   size_t n) {
    for (size_t i = (size_t)blockIdx.x * blockDim.x + threadIdx.x; i < n;
         i += (size_t)gridDim.x * blockDim.x)
        dst[i] = (bf16_t)src[i];
}

__global__ void bias_sum_kernel(const float* __restrict__ a, const float* __restrict__ b,
                                float* __restrict__ o, int n) {
    int i = blockIdx.x * blockDim.x + threadIdx.x;
    if (i < n) o[i] = a[i] + b[i];
}

// Wc[n][k] = k<CE ? Wih_c[n][k] : Whh_c[n][k-CE]   (bf16)
__global__ void build_char_w_kernel(const float* __restrict__ Wih, const float* __restrict__ Whh,
                                    bf16_t* __restrict__ Wc) {
    size_t total = (size_t)NC * KC;
    for (size_t i = (size_t)blockIdx.x * blockDim.x + threadIdx.x; i < total;
         i += (size_t)gridDim.x * blockDim.x) {
        int n = (int)(i / KC), k = (int)(i % KC);
        float v = (k < CE) ? Wih[(size_t)n * CE + k] : Whh[(size_t)n * CE + (k - CE)];
        Wc[i] = (bf16_t)v;
    }
}

// E[t][s][k] = bf16(char_emb[char_idxs[s][t]][k])
__global__ void embed_chars_kernel(const int* __restrict__ char_idxs,
                                   const float* __restrict__ char_emb,
                                   bf16_t* __restrict__ E) {
    size_t total = (size_t)S * LMAX * CE;
    for (size_t i = (size_t)blockIdx.x * blockDim.x + threadIdx.x; i < total;
         i += (size_t)gridDim.x * blockDim.x) {
        int s = (int)(i >> 12);                // / (LMAX*CE)
        int r = (int)(i & 4095);
        int t = r >> 8, k = r & 255;
        int ch = char_idxs[s * LMAX + t];
        E[((size_t)t * S + s) * CE + k] = (bf16_t)char_emb[(size_t)ch * CE + k];
    }
}

// X_w[s][k] = k<WE ? word_emb[word_idxs[s]][k] : last[s][k-WE]   (bf16)
__global__ void pack_xw_kernel(const int* __restrict__ word_idxs,
                               const float* __restrict__ word_emb,
                               const float* __restrict__ last, bf16_t* __restrict__ Xw) {
    size_t total = (size_t)S * KW;
    for (size_t i = (size_t)blockIdx.x * blockDim.x + threadIdx.x; i < total;
         i += (size_t)gridDim.x * blockDim.x) {
        int s = (int)(i / KW), k = (int)(i % KW);
        float v = (k < WE) ? word_emb[(size_t)word_idxs[s] * WE + k]
                           : last[(size_t)s * CE + (k - WE)];
        Xw[i] = (bf16_t)v;
    }
}

// ---------------------------------------------------------------------------
// bf16 WMMA GEMM: C[M][N] = [A0|A1][M][K] * B[N][K]^T + bias[N]
// A row k<K0 -> A0 (ld=K0), else A1 (ld=K-K0). B row-major [N][K] (weights).
// block = 256 thr (8 waves), tile 128x64x32, wave 32x32 (2x2 WMMA accums).
// A/B k-tiles double-buffered in LDS via async load-to-LDS.
// ---------------------------------------------------------------------------
__global__ __launch_bounds__(256) void gemm_bf16_wmma_kernel(
    const bf16_t* __restrict__ A0, const bf16_t* __restrict__ A1, int K0, int K,
    const bf16_t* __restrict__ B, const float* __restrict__ bias,
    float* __restrict__ C, int N) {
    __shared__ __align__(16) bf16_t As[2][128 * 32];   // [row][k] 64B rows
    __shared__ __align__(16) bf16_t Bs[2][64 * 32];    // [col][k] 64B rows

    const int tid  = threadIdx.x;
    const int lane = tid & 31;
    const int wave = tid >> 5;
    const int waveM = wave >> 1;              // 0..3
    const int waveN = wave & 1;               // 0..1
    const int rowBlk = blockIdx.y * 128;
    const int colBlk = blockIdx.x * 64;
    const int lhi = lane >> 4;                // 0/1
    const int llo = lane & 15;

    // staging coords: A = 512 x 16B chunks (2/thread), B = 256 x 16B (1/thread)
    const int ar = tid >> 1;                  // A row 0..127
    const int ap = (tid & 1) * 2;             // A 16B-part 0 or 2 (copy 2 parts)
    const int bn = tid >> 2;                  // B col 0..63
    const int bp = tid & 3;                   // B 16B-part 0..3

    auto stageA = [&](int kt, int buf) {
        const bf16_t* Asrc; int ld, kloc;
        if (kt < K0) { Asrc = A0; ld = K0; kloc = kt; }
        else         { Asrc = A1; ld = K - K0; kloc = kt - K0; }
        const bf16_t* g = Asrc + (size_t)(rowBlk + ar) * ld + kloc + ap * 8;
        bf16_t* l = &As[buf][ar * 32 + ap * 8];
        stage16(g, l);
        stage16(g + 8, l + 8);
    };
    auto stageB = [&](int kt, int buf) {
        const bf16_t* g = B + (size_t)(colBlk + bn) * K + kt + bp * 8;
        stage16(g, &Bs[buf][bn * 32 + bp * 8]);
    };

    v8f acc[2][2] = {};

    stageA(0, 0);
    stageB(0, 0);
    wait_stage();

    int buf = 0;
    for (int kt = 0; kt < K; kt += 32) {
        if (kt + 32 < K) { stageA(kt + 32, buf ^ 1); stageB(kt + 32, buf ^ 1); }

        v16bf afr[2];
#pragma unroll
        for (int mi = 0; mi < 2; ++mi) {
            const bf16_t* p = &As[buf][(waveM * 32 + mi * 16 + llo) * 32 + lhi * 16];
            afr[mi] = comb16(*(const bf16x8*)p, *(const bf16x8*)(p + 8));
        }
        v16bf bfr[2];
#pragma unroll
        for (int ni = 0; ni < 2; ++ni) {
            const bf16_t* p = &Bs[buf][(waveN * 32 + ni * 16 + llo) * 32 + lhi * 16];
            bfr[ni] = comb16(*(const bf16x8*)p, *(const bf16x8*)(p + 8));
        }
#pragma unroll
        for (int mi = 0; mi < 2; ++mi)
#pragma unroll
            for (int ni = 0; ni < 2; ++ni)
                acc[mi][ni] = __builtin_amdgcn_wmma_f32_16x16x32_bf16(
                    false, afr[mi], false, bfr[ni], (short)0, acc[mi][ni], false, false);

        wait_stage();   // drain async tile for buf^1, fence LDS reuse of buf
        buf ^= 1;
    }

#pragma unroll
    for (int mi = 0; mi < 2; ++mi)
#pragma unroll
        for (int ni = 0; ni < 2; ++ni) {
            const int n = colBlk + waveN * 32 + ni * 16 + llo;
            const float bv = bias[n];
#pragma unroll
            for (int v = 0; v < 8; ++v) {
                const int s = rowBlk + waveM * 32 + mi * 16 + v + lhi * 8;
                C[(size_t)s * N + n] = acc[mi][ni][v] + bv;
            }
        }
}

// ---------------------------------------------------------------------------
// char LSTM cell (torch gate order i,f,g,o), writes bf16 h for the next GEMM,
// harvests last hidden state at t == len-1.
// ---------------------------------------------------------------------------
__global__ __launch_bounds__(256) void char_cell_kernel(
    const float* __restrict__ gates, float* __restrict__ c_state,
    bf16_t* __restrict__ h_bf, float* __restrict__ last,
    const int* __restrict__ char_lens, int t) {
    const int s = blockIdx.x, j = threadIdx.x;
    const size_t g0 = (size_t)s * NC;
    const float gi = gates[g0 + j];
    const float gf = gates[g0 + CE + j];
    const float gg = gates[g0 + 2 * CE + j];
    const float go = gates[g0 + 3 * CE + j];
    const size_t idx = (size_t)s * CE + j;
    float c = sigm(gf) * c_state[idx] + sigm(gi) * tanhf(gg);
    c_state[idx] = c;
    const float h = sigm(go) * tanhf(c);
    h_bf[idx] = (bf16_t)h;
    if (t == char_lens[s] - 1) last[idx] = h;
}

// ---------------------------------------------------------------------------
// word LSTM recurrence (batch=1): persistent kernel, 8 blocks x 256 threads.
// Block b owns hidden units j in [b*128, b*128+128) and their 4 gate rows.
// ---------------------------------------------------------------------------
__global__ __launch_bounds__(256) void word_lstm_kernel(
    const float* __restrict__ G_in,      // [S][NW]
    const bf16_t* __restrict__ Whh,      // [NW][HDIM] bf16
    bf16_t* __restrict__ h_bf,           // [HDIM] (zero-initialized)
    float* __restrict__ out,             // [S][HDIM]
    unsigned* __restrict__ counters) {
    __shared__ __align__(16) bf16_t hsh[HDIM];
    __shared__ float gsh[512];
    __shared__ float csh[128];

    const int tid = threadIdx.x;
    const int b = blockIdx.x;
    const unsigned nb = gridDim.x;

    if (tid < 128) csh[tid] = 0.0f;

    for (int t = 0; t < S; ++t) {
        // h_prev -> LDS (4 bf16 per thread)
        *(uint2*)(&hsh[tid * 4]) = *(const uint2*)(h_bf + tid * 4);
        __syncthreads();

        // prefetch next step's gate-input row segment into L2
        if (t + 1 < S) __builtin_prefetch(G_in + (size_t)(t + 1) * NW + b * 128, 0, 1);

#pragma unroll
        for (int oo = 0; oo < 2; ++oo) {
            const int o = tid + oo * 256;         // 0..511
            const int sec = o >> 7, jj = o & 127;
            const int n = sec * HDIM + b * 128 + jj;
            const bf16_t* wrow = Whh + (size_t)n * HDIM;
            float acc = G_in[(size_t)t * NW + n];
            for (int k = 0; k < HDIM; k += 8) {
                bf16x8 w = *(const bf16x8*)(wrow + k);
                bf16x8 h = *(const bf16x8*)(&hsh[k]);
#pragma unroll
                for (int i = 0; i < 8; ++i) acc += (float)w[i] * (float)h[i];
            }
            gsh[o] = acc;
        }
        __syncthreads();

        if (tid < 128) {
            const float gi = gsh[tid];
            const float gf = gsh[128 + tid];
            const float gg = gsh[256 + tid];
            const float go = gsh[384 + tid];
            float c = sigm(gf) * csh[tid] + sigm(gi) * tanhf(gg);
            csh[tid] = c;
            const float h = sigm(go) * tanhf(c);
            out[(size_t)t * HDIM + b * 128 + tid] = h;
            h_bf[b * 128 + tid] = (bf16_t)h;
        }
        __syncthreads();

        // device-wide barrier: one arrival per block per step
        if (tid == 0) {
            __threadfence();
            atomicAdd(&counters[0], 1u);
            const unsigned target = nb * (unsigned)(t + 1);
            while (__hip_atomic_load(&counters[0], __ATOMIC_ACQUIRE,
                                     __HIP_MEMORY_SCOPE_AGENT) < target)
                __builtin_amdgcn_s_sleep(2);
        }
        __syncthreads();
    }
}

// ---------------------------------------------------------------------------
extern "C" void kernel_launch(void* const* d_in, const int* in_sizes, int n_in,
                              void* d_out, int out_size, void* d_ws, size_t ws_size,
                              hipStream_t stream) {
    (void)in_sizes; (void)n_in; (void)out_size; (void)ws_size;
    const int*   word_idxs = (const int*)d_in[0];
    const int*   char_idxs = (const int*)d_in[1];
    const int*   char_lens = (const int*)d_in[2];
    const float* char_emb  = (const float*)d_in[3];
    const float* word_emb  = (const float*)d_in[4];
    const float* Wih_c     = (const float*)d_in[5];
    const float* Whh_c     = (const float*)d_in[6];
    const float* bih_c     = (const float*)d_in[7];
    const float* bhh_c     = (const float*)d_in[8];
    const float* Wih_w     = (const float*)d_in[9];
    const float* Whh_w     = (const float*)d_in[10];
    const float* bih_w     = (const float*)d_in[11];
    const float* bhh_w     = (const float*)d_in[12];

    char* ws = (char*)d_ws;
    bf16_t*   Hbf     = (bf16_t*)(ws + OFF_HBF);
    float*    Cc      = (float*)(ws + OFF_CC);
    bf16_t*   Hw      = (bf16_t*)(ws + OFF_HW);
    unsigned* counters= (unsigned*)(ws + OFF_CTR);
    bf16_t*   E       = (bf16_t*)(ws + OFF_E);
    bf16_t*   Wc      = (bf16_t*)(ws + OFF_WC);
    float*    bias_c  = (float*)(ws + OFF_BC);
    float*    last    = (float*)(ws + OFF_LAST);
    float*    gates   = (float*)(ws + OFF_GATES);
    bf16_t*   Xw      = (bf16_t*)(ws + OFF_XW);
    bf16_t*   WihWbf  = (bf16_t*)(ws + OFF_WIHW);
    bf16_t*   WhhWbf  = (bf16_t*)(ws + OFF_WHHW);
    float*    bias_w  = (float*)(ws + OFF_BW);
    float*    Gin     = (float*)(ws + OFF_GIN);
    float*    out     = (float*)d_out;

    // zero recurrent state + barrier counters (must be fresh every launch)
    zero_ws_kernel<<<256, 256, 0, stream>>>((uint4*)d_ws, ZERO_BYTES / 16);

    // weight / bias / embedding prep
    bias_sum_kernel<<<(NC + 255) / 256, 256, 0, stream>>>(bih_c, bhh_c, bias_c, NC);
    bias_sum_kernel<<<(NW + 255) / 256, 256, 0, stream>>>(bih_w, bhh_w, bias_w, NW);
    build_char_w_kernel<<<1024, 256, 0, stream>>>(Wih_c, Whh_c, Wc);
    f32_to_bf16_kernel<<<2048, 256, 0, stream>>>(Wih_w, WihWbf, (size_t)NW * KW);
    f32_to_bf16_kernel<<<2048, 256, 0, stream>>>(Whh_w, WhhWbf, (size_t)NW * HDIM);
    embed_chars_kernel<<<4096, 256, 0, stream>>>(char_idxs, char_emb, E);

    // char LSTM: 16 dependent WMMA GEMM + cell steps
    for (int t = 0; t < LMAX; ++t) {
        gemm_bf16_wmma_kernel<<<dim3(NC / 64, S / 128), 256, 0, stream>>>(
            E + (size_t)t * S * CE, Hbf, CE, KC, Wc, bias_c, gates, NC);
        char_cell_kernel<<<S, 256, 0, stream>>>(gates, Cc, Hbf, last, char_lens, t);
    }

    // word LSTM input projection (batched WMMA GEMM)
    pack_xw_kernel<<<6144, 256, 0, stream>>>(word_idxs, word_emb, last, Xw);
    gemm_bf16_wmma_kernel<<<dim3(NW / 64, S / 128), 256, 0, stream>>>(
        Xw, Xw, KW, KW, WihWbf, bias_w, Gin, NW);

    // word LSTM recurrence (persistent, device-wide barrier per step)
    word_lstm_kernel<<<8, 256, 0, stream>>>(Gin, WhhWbf, Hw, out, counters);
}
